// InitStr_Network_60790967108020
// MI455X (gfx1250) — compile-verified, matrix operand order checked by reading, not run on hardware
//
#include <hip/hip_runtime.h>
#include <hip/hip_bf16.h>
#include <hip/hip_fp16.h>

typedef __attribute__((ext_vector_type(16))) _Float16 v16h;
typedef __attribute__((ext_vector_type(8)))  float    v8f;

#define L_ 384
#define NSEQ 128
#define DH 64
#define HD 256
#define NEGBIG -1e9f

// ---------------- param leaf indices (jax pytree: dict keys sorted, lists in order) ----
#define P0 4
#define BLK(i) (P0 + (i)*13)
#define B_BK 0
#define B_BOUT 1
#define B_BQ 2
#define B_BSKIP 3
#define B_BV 4
#define B_LNB 5
#define B_LNG 6
#define B_WE 7
#define B_WK 8
#define B_WOUT 9
#define B_WQ 10
#define B_WSKIP 11
#define B_WV 12
#define IEE_B   (P0+39)
#define IEE_LNB (P0+40)
#define IEE_LNG (P0+41)
#define IEE_W   (P0+42)
#define IEX_B   (P0+43)
#define IEX_LNB (P0+44)
#define IEX_LNG (P0+45)
#define IEX_W   (P0+46)
#define IL1_B   (P0+47)
#define IL1_W   (P0+48)
#define IGS_B   (P0+49)
#define IGS_W   (P0+50)
#define INE_B   (P0+51)
#define INE_G   (P0+52)
#define INN_B   (P0+53)
#define INN_G   (P0+54)
#define INS_B   (P0+55)
#define INS_G   (P0+56)
#define ISW_KB  (P0+57)
#define ISW_KW  (P0+58)
#define ISW_QB  (P0+59)
#define ISW_QW  (P0+60)

// ---- WMMA 16-bit fragment element<->K mapping (CDNA5 ISA 7.12.2) ----
__device__ __forceinline__ int wmma_k_of(int e, int lane) {
  return ((e < 8) ? e : (e + 8)) + ((lane < 16) ? 0 : 8);
}
// inverse: source k (0..31) -> (hi half-wave?, element)
__device__ __forceinline__ void wmma_inv_k(int k, int& hi, int& e) {
  if (k < 8)       { hi = 0; e = k; }
  else if (k < 16) { hi = 1; e = k - 8; }
  else if (k < 24) { hi = 0; e = k - 8; }
  else             { hi = 1; e = k - 16; }
}

// ---------------- prep / packing kernels ----------------
__global__ void cvt_f16_kernel(_Float16* __restrict__ dst, const float* __restrict__ src, int n) {
  int i = blockIdx.x * blockDim.x + threadIdx.x;
  if (i < n) dst[i] = (_Float16)src[i];
}
// src R x C f32 -> dst C x R f16 (transpose)
__global__ void transpose_cvt_kernel(_Float16* __restrict__ dst, const float* __restrict__ src, int R, int C) {
  int i = blockIdx.x * blockDim.x + threadIdx.x;
  if (i < R * C) { int r = i / C, c = i % C; dst[c * R + r] = (_Float16)src[i]; }
}
// Pack A (row-major f32, leading dim lda) into fragment-ready f16:
// dst[((mt*nk + kc)*32 + lane)*16 + e] = A[mt*16 + (lane&15)][kc*32 + k_of(e,lane)]
__global__ void pack_a_kernel(_Float16* __restrict__ dst, const float* __restrict__ src,
                              int lda, int M, int nk) {
  int i = blockIdx.x * blockDim.x + threadIdx.x;
  int total = M * nk * 32;  // halves/16
  if (i >= total * 16) return;
  int e = i & 15, lane = (i >> 4) & 31, kc = (i >> 9) % nk, mt = i >> 9; mt /= nk;
  int row = mt * 16 + (lane & 15);
  int k = kc * 32 + wmma_k_of(e, lane);
  dst[i] = (_Float16)src[(size_t)row * lda + k];
}
// Pack B (row-major f32 KxN, ldb==N) into fragment-ready f16:
// dst[((ct*nk + kc)*32 + lane)*16 + e] = B[kc*32 + k_of(e,lane)][ct*16 + (lane&15)]
__global__ void pack_b_kernel(_Float16* __restrict__ dst, const float* __restrict__ src,
                              int ldb, int N, int nk) {
  int i = blockIdx.x * blockDim.x + threadIdx.x;
  if (i >= (N / 16) * nk * 512) return;
  int e = i & 15, lane = (i >> 4) & 31, kc = (i >> 9) % nk, ct = (i >> 9) / nk;
  int col = ct * 16 + (lane & 15);
  int k = kc * 32 + wmma_k_of(e, lane);
  dst[i] = (_Float16)src[(size_t)k * ldb + col];
}
// Pack k^T per head straight from kbuf f32 (L_ x 256): B_h[k][s] = kbuf[s*256 + h*64 + k]
// per-head packed size = (384/16)*2*512 = 24576
__global__ void pack_kT_kernel(_Float16* __restrict__ dst, const float* __restrict__ kbuf) {
  int i = blockIdx.x * blockDim.x + threadIdx.x;
  if (i >= 4 * 24576) return;
  int h = i / 24576, r = i % 24576;
  int e = r & 15, lane = (r >> 4) & 31, kc = (r >> 9) & 1, ct = r >> 10;
  int s = ct * 16 + (lane & 15);
  int k = kc * 32 + wmma_k_of(e, lane);
  dst[i] = (_Float16)kbuf[(size_t)s * HD + h * 64 + k];
}
// edge-embed weight (129 x 64) -> packed padded (K=160 -> nk=5, N=64 -> nct=4)
__global__ void prep_wee_kernel(_Float16* __restrict__ dst, const float* __restrict__ w) {
  int i = blockIdx.x * blockDim.x + threadIdx.x;
  if (i >= 4 * 5 * 512) return;
  int e = i & 15, lane = (i >> 4) & 31, kc = (i >> 9) % 5, ct = (i >> 9) / 5;
  int col = ct * 16 + (lane & 15);
  int kk = kc * 32 + wmma_k_of(e, lane);
  dst[i] = (kk < 129) ? (_Float16)w[kk * 64 + col] : (_Float16)0.f;
}

// ---------------- fused seq-attention + node embed (per residue l) ----------------
__global__ void seq_embed_kernel(const float* __restrict__ msa, const float* __restrict__ seq1hot,
                                 const float* __restrict__ nng, const float* __restrict__ nnb,
                                 const float* __restrict__ qw, const float* __restrict__ qb,
                                 const float* __restrict__ kw, const float* __restrict__ kb,
                                 const float* __restrict__ exw, const float* __restrict__ exb,
                                 const float* __restrict__ exlg, const float* __restrict__ exlb,
                                 float* __restrict__ x0) {
  __shared__ float mln[NSEQ * DH];
  __shared__ float qs[DH];
  __shared__ float attn[NSEQ];
  __shared__ float wn[NSEQ];
  __shared__ float nodev[85];
  __shared__ float ybuf[DH];
  const int l = blockIdx.x, tid = threadIdx.x;
  {
    const float* p = msa + ((size_t)tid * L_ + l) * DH;
    float mu = 0.f; for (int c = 0; c < DH; ++c) mu += p[c]; mu *= (1.f / DH);
    float var = 0.f; for (int c = 0; c < DH; ++c) { float d = p[c] - mu; var += d * d; } var *= (1.f / DH);
    float rs = rsqrtf(var + 1e-5f);
    float* o = mln + tid * DH;
    for (int c = 0; c < DH; ++c) o[c] = nng[c] * (p[c] - mu) * rs + nnb[c];
  }
  __syncthreads();
  if (tid < DH) {
    float s = qb[tid];
    for (int j = 0; j < DH; ++j) s += mln[j] * qw[j * DH + tid];
    qs[tid] = s;
  }
  __syncthreads();
  {
    const float* m = mln + tid * DH;
    float acc = 0.f;
    for (int k = 0; k < DH; ++k) {
      float s = kb[k];
      for (int j = 0; j < DH; ++j) s += m[j] * kw[j * DH + k];
      acc += s * qs[k];
    }
    attn[tid] = acc * 0.125f;
  }
  __syncthreads();
  {
    float m = -3.4e38f; for (int n = 0; n < NSEQ; ++n) m = fmaxf(m, attn[n]);
    float sum = 0.f;    for (int n = 0; n < NSEQ; ++n) sum += expf(attn[n] - m);
    wn[tid] = expf(attn[tid] - m) / sum;
  }
  __syncthreads();
  if (tid < DH) {
    float a = 0.f;
    for (int n = 0; n < NSEQ; ++n) a += wn[n] * mln[n * DH + tid];
    nodev[tid] = a;
  }
  if (tid >= DH && tid < 85) nodev[tid] = seq1hot[l * 21 + (tid - DH)];
  __syncthreads();
  if (tid < DH) {
    float y = exb[tid];
    for (int j = 0; j < 85; ++j) y += nodev[j] * exw[j * DH + tid];
    ybuf[tid] = y;
  }
  __syncthreads();
  if (tid < DH) {
    float mu = 0.f, sq = 0.f;
    for (int j = 0; j < DH; ++j) { mu += ybuf[j]; sq += ybuf[j] * ybuf[j]; }
    mu *= (1.f / DH); float var = sq * (1.f / DH) - mu * mu;
    float rs = rsqrtf(var + 1e-5f);
    x0[l * DH + tid] = exlg[tid] * (ybuf[tid] - mu) * rs + exlb[tid];
  }
}

// ---------------- fused LN -> WMMA GEMM -> LN edge embed; 16 (s,t)-pairs per block -----
// A-tile written to LDS in fragment-swizzled order -> reads are 2x ds_load_b128.
// Weight is pre-packed in global -> reads are 2x global_load_b128 per chunk.
__global__ void embed_e_kernel(const float* __restrict__ pair, const int* __restrict__ idx,
                               const float* __restrict__ neg, const float* __restrict__ neb,
                               const _Float16* __restrict__ WeePk, const float* __restrict__ eeb,
                               const float* __restrict__ eelg, const float* __restrict__ eelb,
                               _Float16* __restrict__ ET) {
  __shared__ _Float16 Asw[5 * 512];   // (kc, lane, e)
  __shared__ float    Cs[16 * 64];
  const int tid = threadIdx.x, lane = tid & 31, wave = tid >> 5;
  if (tid < 16) {  // row LN of pair[i,j,:128] + seqsep + zero pad, swizzled store
    int r = blockIdx.x * 16 + tid;
    int i = r / L_, j = r % L_;
    const float* p = pair + (size_t)r * 128;
    float mu = 0.f; for (int c = 0; c < 128; ++c) mu += p[c]; mu *= (1.f / 128.f);
    float var = 0.f; for (int c = 0; c < 128; ++c) { float d = p[c] - mu; var += d * d; } var *= (1.f / 128.f);
    float rs = rsqrtf(var + 1e-5f);
    int sep = idx[j] - idx[i];
    float ssv = (sep == 1) ? 1.f : ((sep == -1) ? -1.f : 0.f);
    for (int c = 0; c < 160; ++c) {
      float v = (c < 128) ? (neg[c] * (p[c] - mu) * rs + neb[c]) : ((c == 128) ? ssv : 0.f);
      int kc = c >> 5, kk = c & 31, hi, e;
      wmma_inv_k(kk, hi, e);
      Asw[((kc * 32) + tid + (hi ? 16 : 0)) * 16 + e] = (_Float16)v;
    }
  }
  __syncthreads();
  v8f acc = {};
#pragma unroll
  for (int kc = 0; kc < 5; ++kc) {
    v16h a = *(const v16h*)(Asw + (kc * 32 + lane) * 16);
    v16h b = *(const v16h*)(WeePk + (((size_t)wave * 5 + kc) * 32 + lane) * 16);
    acc = __builtin_amdgcn_wmma_f32_16x16x32_f16(false, a, false, b, (short)0, acc, false, false);
  }
  {
    int col = wave * 16 + (lane & 15);
    float bv = eeb[col];
#pragma unroll
    for (int r8 = 0; r8 < 8; ++r8) {
      int row = r8 + ((lane < 16) ? 0 : 8);
      Cs[row * 64 + col] = acc[r8] + bv;
    }
  }
  __syncthreads();
  if (tid < 16) {  // output LN, store transposed: ET[(j*384+i)*64 + c] = e_emb[i,j,c]
    int r = blockIdx.x * 16 + tid;
    int i = r / L_, j = r % L_;
    const float* y = Cs + tid * 64;
    float mu = 0.f, sq = 0.f;
    for (int c = 0; c < 64; ++c) { mu += y[c]; sq += y[c] * y[c]; }
    mu *= (1.f / 64.f); float var = sq * (1.f / 64.f) - mu * mu;
    float rs = rsqrtf(var + 1e-5f);
    _Float16* o = ET + ((size_t)j * L_ + i) * 64;
    for (int c = 0; c < 64; ++c) o[c] = (_Float16)(eelg[c] * (y[c] - mu) * rs + eelb[c]);
  }
}

// ---------------- LDS-free WMMA GEMM, K=64, pre-packed A/B --------------------
// grid = (M/16, N/64), block = 128 (4 waves, one 16x16 tile each).
__global__ void gemm_wmma64_kernel(const _Float16* __restrict__ Ap, const _Float16* __restrict__ Bp,
                                   const float* __restrict__ bias, float* __restrict__ C, int ldc) {
  const int tid = threadIdx.x, lane = tid & 31, wave = tid >> 5;
  const int mt = blockIdx.x;
  const int ct = blockIdx.y * 4 + wave;
  const _Float16* a0 = Ap + ((size_t)mt * 2) * 512 + lane * 16;
  const _Float16* b0 = Bp + ((size_t)ct * 2) * 512 + lane * 16;
  v8f acc = {};
#pragma unroll
  for (int kc = 0; kc < 2; ++kc) {
    v16h a = *(const v16h*)(a0 + kc * 512);
    v16h b = *(const v16h*)(b0 + kc * 512);
    acc = __builtin_amdgcn_wmma_f32_16x16x32_f16(false, a, false, b, (short)0, acc, false, false);
  }
  int col = ct * 16 + (lane & 15);
  float bv = bias ? bias[col] : 0.f;
#pragma unroll
  for (int r = 0; r < 8; ++r) {
    int row = mt * 16 + r + ((lane < 16) ? 0 : 8);
    C[(size_t)row * ldc + col] = acc[r] + bv;
  }
}

// ---------------- fused per-target attention block (edge terms refactored) --------------
__global__ void block_attn_kernel(const float* __restrict__ x, const float* __restrict__ q,
                                  const float* __restrict__ v, const float* __restrict__ skip,
                                  const float* __restrict__ lqk, const _Float16* __restrict__ ET,
                                  const _Float16* __restrict__ weT, const _Float16* __restrict__ we,
                                  const float* __restrict__ lng, const float* __restrict__ lnb,
                                  const _Float16* __restrict__ wout, const float* __restrict__ bout,
                                  float* __restrict__ x_out) {
  __shared__ float qrow[HD], qe[HD], lg[L_ * 4], ebar[HD], agg[HD], h256[HD];
  __shared__ float mx[4], dn[4];
  const int t = blockIdx.x, tid = threadIdx.x;
  qrow[tid] = q[t * HD + tid];
  __syncthreads();
  {  // qe[h,c] = sum_d q[t,h,d] * we[c, h*64+d]
    int h = tid >> 6, c = tid & 63;
    const _Float16* wp = weT + (size_t)(h * 64) * 64 + c;
    float s = 0.f;
    for (int d = 0; d < 64; ++d) s += qrow[h * 64 + d] * (float)wp[d * 64];
    qe[tid] = s;
  }
  __syncthreads();
  for (int it = 0; it < 6; ++it) {
    int item = tid + it * 256;
    int s = item >> 2, h = item & 3;
    const _Float16* ep = ET + ((size_t)t * L_ + s) * 64;
    if (s + 4 < L_) __builtin_prefetch(ET + ((size_t)t * L_ + s + 4) * 64, 0, 3);
    float le = 0.f;
    for (int c = 0; c < 64; ++c) le += qe[h * 64 + c] * (float)ep[c];
    float Lg = (lqk[(size_t)h * L_ * L_ + t * L_ + s] + le) * 0.125f;
    if (s == t) Lg = NEGBIG;
    lg[s * 4 + h] = Lg;
  }
  __syncthreads();
  if (tid < 4) {
    float m = -3.4e38f; for (int s = 0; s < L_; ++s) m = fmaxf(m, lg[s * 4 + tid]);
    float sum = 0.f;    for (int s = 0; s < L_; ++s) sum += expf(lg[s * 4 + tid] - m);
    mx[tid] = m; dn[tid] = 1.f / sum;
  }
  __syncthreads();
  for (int it = 0; it < 6; ++it) {
    int item = tid + it * 256;
    int s = item >> 2, h = item & 3;
    lg[s * 4 + h] = expf(lg[s * 4 + h] - mx[h]) * dn[h];
  }
  __syncthreads();
  {
    int h = tid >> 6, d = tid & 63;
    float av = 0.f, ae = 0.f;
    for (int s = 0; s < L_; ++s) {
      float a = lg[s * 4 + h];
      av += a * v[(size_t)s * HD + h * 64 + d];
      ae += a * (float)ET[((size_t)t * L_ + s) * 64 + d];
    }
    ebar[tid] = ae;
    agg[tid] = av;
  }
  __syncthreads();
  {
    int h = tid >> 6, d = tid & 63;
    float s2 = 0.f;
    for (int c = 0; c < 64; ++c) s2 += ebar[h * 64 + c] * (float)we[c * HD + h * 64 + d];
    agg[tid] += s2 + skip[t * HD + tid];
  }
  __syncthreads();
  {
    float mu = 0.f, sq = 0.f;
    for (int j = 0; j < HD; ++j) { float y = agg[j]; mu += y; sq += y * y; }
    mu *= (1.f / HD); float var = sq * (1.f / HD) - mu * mu;
    float rs = rsqrtf(var + 1e-5f);
    h256[tid] = lng[tid] * (agg[tid] - mu) * rs + lnb[tid];
  }
  __syncthreads();
  if (tid < 64) {
    float o = bout[tid];
    for (int j = 0; j < HD; ++j) o += h256[j] * (float)wout[j * 64 + tid];
    float z = o + x[t * DH + tid];
    x_out[t * DH + tid] = (z > 0.f) ? z : (expf(z) - 1.f);
  }
}

// ---------------- final coordinate + state heads ----------------
__global__ void head_kernel(const float* __restrict__ x,
                            const float* __restrict__ l1w, const float* __restrict__ l1b,
                            const float* __restrict__ nsg, const float* __restrict__ nsb,
                            const float* __restrict__ gsw, const float* __restrict__ gsb,
                            float* __restrict__ out) {
  __shared__ float xr[64], lnr[64], l1[6];
  const int t = blockIdx.x, tid = threadIdx.x;
  xr[tid] = x[t * 64 + tid];
  __syncthreads();
  if (tid < 6) {
    float s = l1b[tid];
    for (int d = 0; d < 64; ++d) s += xr[d] * l1w[d * 6 + tid];
    l1[tid] = s;
  }
  {
    float mu = 0.f, sq = 0.f;
    for (int d = 0; d < 64; ++d) { mu += xr[d]; sq += xr[d] * xr[d]; }
    mu *= (1.f / 64.f); float var = sq * (1.f / 64.f) - mu * mu;
    float rs = rsqrtf(var + 1e-5f);
    lnr[tid] = nsg[tid] * (xr[tid] - mu) * rs + nsb[tid];
  }
  __syncthreads();
  if (tid < 8) {
    float s = gsb[tid];
    for (int d = 0; d < 64; ++d) s += lnr[d] * gsw[d * 8 + tid];
    out[L_ * 9 + t * 8 + tid] = s;
  }
  if (tid == 0) {
    float T0 = l1[0], T1 = l1[1], T2 = l1[2];
    float R0 = l1[3], R1 = l1[4], R2 = l1[5];
    float ang = sqrtf(R0 * R0 + R1 * R1 + R2 * R2);
    float inv = 1.f / (ang + 1e-5f);
    float r0 = R0 * inv, r1 = R1 * inv, r2 = R2 * inv;
    float ca = cosf(ang), sa = sinf(ang);
    const float V[3][3] = {{-0.5272f, 1.3593f, 0.f}, {0.f, 0.f, 0.f}, {1.5233f, 0.f, 0.f}};
    for (int a = 0; a < 3; ++a) {
      float v0 = V[a][0], v1 = V[a][1], v2 = V[a][2];
      float rd = r0 * v0 + r1 * v1 + r2 * v2;
      float cx = r1 * v2 - r2 * v1, cy = r2 * v0 - r0 * v2, cz = r0 * v1 - r1 * v0;
      out[t * 9 + a * 3 + 0] = (v0 - r0 * rd) * ca + cx * sa + r0 * rd + T0;
      out[t * 9 + a * 3 + 1] = (v1 - r1 * rd) * ca + cy * sa + r1 * rd + T1;
      out[t * 9 + a * 3 + 2] = (v2 - r2 * rd) * ca + cz * sa + r2 * rd + T2;
    }
  }
}

// ---------------- host orchestration ----------------
extern "C" void kernel_launch(void* const* d_in, const int* in_sizes, int n_in,
                              void* d_out, int out_size, void* d_ws, size_t ws_size,
                              hipStream_t stream) {
  (void)in_sizes; (void)n_in; (void)out_size; (void)ws_size;
  const float* seq1hot = (const float*)d_in[0];
  const int*   idx     = (const int*)d_in[1];
  const float* msa     = (const float*)d_in[2];
  const float* pair    = (const float*)d_in[3];
  float* out = (float*)d_out;

  char* p = (char*)d_ws;
  auto alloc = [&](size_t bytes) -> void* {
    void* r = (void*)p; p += (bytes + 255) & ~(size_t)255; return r;
  };
  _Float16* ET    = (_Float16*)alloc((size_t)L_ * L_ * 64 * 2);
  float*    xa    = (float*)alloc(L_ * DH * 4);
  float*    xb    = (float*)alloc(L_ * DH * 4);
  _Float16* xPk   = (_Float16*)alloc(L_ * DH * 2);       // packed A, 24 mt * 2 kc * 512
  _Float16* qPk   = (_Float16*)alloc(4 * 24576 * 2);     // packed A per head
  _Float16* kTPk  = (_Float16*)alloc(4 * 24576 * 2);     // packed B per head
  float*    qbuf  = (float*)alloc(L_ * HD * 4);
  float*    kbuf  = (float*)alloc(L_ * HD * 4);
  float*    vbuf  = (float*)alloc(L_ * HD * 4);
  float*    sbuf  = (float*)alloc(L_ * HD * 4);
  float*    lqk   = (float*)alloc((size_t)4 * L_ * L_ * 4);
  _Float16* WeePk = (_Float16*)alloc(4 * 5 * 512 * 2);
  _Float16 *wqP[3], *wkP[3], *wvP[3], *wsP[3], *wef[3], *weTf[3], *wouf[3];
  for (int b = 0; b < 3; ++b) {
    wqP[b]  = (_Float16*)alloc(16 * 2 * 512 * 2);  // N/16=16 ct, nk=2
    wkP[b]  = (_Float16*)alloc(16 * 2 * 512 * 2);
    wvP[b]  = (_Float16*)alloc(16 * 2 * 512 * 2);
    wsP[b]  = (_Float16*)alloc(16 * 2 * 512 * 2);
    wef[b]  = (_Float16*)alloc(64 * 256 * 2);
    weTf[b] = (_Float16*)alloc(256 * 64 * 2);
    wouf[b] = (_Float16*)alloc(256 * 64 * 2);
  }

  auto cvt = [&](_Float16* d, const float* s, int n) {
    cvt_f16_kernel<<<(n + 255) / 256, 256, 0, stream>>>(d, s, n);
  };

  // ---- weight prep ----
  prep_wee_kernel<<<(4 * 5 * 512 + 255) / 256, 256, 0, stream>>>(WeePk, (const float*)d_in[IEE_W]);
  for (int b = 0; b < 3; ++b) {
    const int nB = 16 * 2 * 512;
    pack_b_kernel<<<(nB + 255) / 256, 256, 0, stream>>>(wqP[b], (const float*)d_in[BLK(b) + B_WQ], HD, HD, 2);
    pack_b_kernel<<<(nB + 255) / 256, 256, 0, stream>>>(wkP[b], (const float*)d_in[BLK(b) + B_WK], HD, HD, 2);
    pack_b_kernel<<<(nB + 255) / 256, 256, 0, stream>>>(wvP[b], (const float*)d_in[BLK(b) + B_WV], HD, HD, 2);
    pack_b_kernel<<<(nB + 255) / 256, 256, 0, stream>>>(wsP[b], (const float*)d_in[BLK(b) + B_WSKIP], HD, HD, 2);
    cvt(wef[b], (const float*)d_in[BLK(b) + B_WE], 64 * 256);
    transpose_cvt_kernel<<<(64 * 256 + 255) / 256, 256, 0, stream>>>(
        weTf[b], (const float*)d_in[BLK(b) + B_WE], 64, 256);
    cvt(wouf[b], (const float*)d_in[BLK(b) + B_WOUT], 256 * 64);
  }

  // ---- seq attention + node embed -> xa ----
  seq_embed_kernel<<<L_, 128, 0, stream>>>(
      msa, seq1hot,
      (const float*)d_in[INN_G], (const float*)d_in[INN_B],
      (const float*)d_in[ISW_QW], (const float*)d_in[ISW_QB],
      (const float*)d_in[ISW_KW], (const float*)d_in[ISW_KB],
      (const float*)d_in[IEX_W], (const float*)d_in[IEX_B],
      (const float*)d_in[IEX_LNG], (const float*)d_in[IEX_LNB], xa);

  // ---- edge embed (LN -> WMMA -> LN) -> ET ----
  embed_e_kernel<<<(L_ * L_) / 16, 128, 0, stream>>>(
      pair, idx,
      (const float*)d_in[INE_G], (const float*)d_in[INE_B],
      WeePk, (const float*)d_in[IEE_B],
      (const float*)d_in[IEE_LNG], (const float*)d_in[IEE_LNB], ET);

  // ---- transformer blocks ----
  float* xc = xa; float* xn = xb;
  for (int b = 0; b < 3; ++b) {
    pack_a_kernel<<<(L_ * DH + 255) / 256, 256, 0, stream>>>(xPk, xc, DH, L_, 2);
    dim3 gp(L_ / 16, HD / 64);
    gemm_wmma64_kernel<<<gp, 128, 0, stream>>>(xPk, wqP[b], (const float*)d_in[BLK(b) + B_BQ], qbuf, HD);
    gemm_wmma64_kernel<<<gp, 128, 0, stream>>>(xPk, wkP[b], (const float*)d_in[BLK(b) + B_BK], kbuf, HD);
    gemm_wmma64_kernel<<<gp, 128, 0, stream>>>(xPk, wvP[b], (const float*)d_in[BLK(b) + B_BV], vbuf, HD);
    gemm_wmma64_kernel<<<gp, 128, 0, stream>>>(xPk, wsP[b], (const float*)d_in[BLK(b) + B_BSKIP], sbuf, HD);
    for (int h = 0; h < 4; ++h)
      pack_a_kernel<<<(24576 + 255) / 256, 256, 0, stream>>>(qPk + h * 24576, qbuf + h * 64, HD, L_, 2);
    pack_kT_kernel<<<(4 * 24576 + 255) / 256, 256, 0, stream>>>(kTPk, kbuf);
    dim3 gl(L_ / 16, L_ / 64);
    for (int h = 0; h < 4; ++h)
      gemm_wmma64_kernel<<<gl, 128, 0, stream>>>(qPk + h * 24576, kTPk + h * 24576,
          nullptr, lqk + (size_t)h * L_ * L_, L_);
    block_attn_kernel<<<L_, 256, 0, stream>>>(
        xc, qbuf, vbuf, sbuf, lqk, ET, weTf[b], wef[b],
        (const float*)d_in[BLK(b) + B_LNG], (const float*)d_in[BLK(b) + B_LNB],
        wouf[b], (const float*)d_in[BLK(b) + B_BOUT], xn);
    float* tmp = xc; xc = xn; xn = tmp;
  }

  // ---- heads ----
  head_kernel<<<L_, 64, 0, stream>>>(
      xc, (const float*)d_in[IL1_W], (const float*)d_in[IL1_B],
      (const float*)d_in[INS_G], (const float*)d_in[INS_B],
      (const float*)d_in[IGS_W], (const float*)d_in[IGS_B], out);
}
